// LogisticRegressionBaseline_59124519797164
// MI455X (gfx1250) — matrix operand order repaired and checked by reading, not compiled, not run on hardware
//
#include <hip/hip_runtime.h>

typedef __attribute__((ext_vector_type(16))) _Float16 v16h;
typedef __attribute__((ext_vector_type(8)))  _Float16 v8h;
typedef __attribute__((ext_vector_type(8)))  float    v8f;

#define Bb    16
#define Ll    4096
#define Vv    256
#define MROWS (Bb * (Ll - 1))   // 65520, divisible by 16
#define CHUNK 64
#define NCH   (Ll / CHUNK)      // 64 chunks per batch row

// ---------------------------------------------------------------------------
// Pass 1: per-chunk max over time, all V lanes.  grid = B*NCH blocks x 256 thr
// ---------------------------------------------------------------------------
__global__ __launch_bounds__(256)
void chunkmax_kernel(const float* __restrict__ codes, float* __restrict__ cmx) {
    const int blk = blockIdx.x;
    const int b = blk / NCH;
    const int c = blk % NCH;
    const int v = threadIdx.x;
    float m = -__builtin_inff();
    const size_t base = ((size_t)b * Ll + (size_t)c * CHUNK) * Vv + v;
    for (int t = 0; t < CHUNK; ++t)
        m = fmaxf(m, codes[base + (size_t)t * Vv]);
    cmx[((size_t)b * NCH + c) * Vv + v] = m;
}

// ---------------------------------------------------------------------------
// Pass 2: exclusive prefix-max over chunk carries (in place).  4096 threads.
// ---------------------------------------------------------------------------
__global__ __launch_bounds__(256)
void carry_scan_kernel(float* __restrict__ cmx) {
    const int tid = blockIdx.x * 256 + threadIdx.x;   // [0, 4096)
    const int b = tid >> 8;
    const int v = tid & 255;
    float running = -__builtin_inff();
    for (int c = 0; c < NCH; ++c) {
        const size_t idx = ((size_t)b * NCH + c) * Vv + v;
        const float t = cmx[idx];
        cmx[idx] = running;
        running = fmaxf(running, t);
    }
}

// ---------------------------------------------------------------------------
// Pass 3: in-chunk running max with carry-in; emit prefix-max as f16.
// pm is [MROWS][Vv] f16, pm[b*(L-1)+t][v] = max(codes[b, 0..t, v]).
// ---------------------------------------------------------------------------
__global__ __launch_bounds__(256)
void prefix_kernel(const float* __restrict__ codes, const float* __restrict__ cmx,
                   _Float16* __restrict__ pm) {
    const int blk = blockIdx.x;
    const int b = blk / NCH;
    const int c = blk % NCH;
    const int v = threadIdx.x;
    float running = cmx[((size_t)b * NCH + c) * Vv + v];
    const int t0 = c * CHUNK;
    const size_t src = ((size_t)b * Ll + t0) * Vv + v;
    const size_t dst = ((size_t)b * (Ll - 1) + t0) * Vv + v;
    for (int t = 0; t < CHUNK; ++t) {
        running = fmaxf(running, codes[src + (size_t)t * Vv]);
        if (t0 + t < Ll - 1)
            pm[dst + (size_t)t * Vv] = (_Float16)running;
    }
}

// ---------------------------------------------------------------------------
// Pack W[o][v] (fp32, row-major) into f16 B-fragment lane order:
//   wp[kc][nt][lane][e] = W[nt*16 + lane%16][kc*32 + (lane/16)*8 + e + (e>=8 ? 8 : 0)]
// One 32B store per thread per fragment; GEMM reads each fragment with a
// single coalesced v16h load.  4096 threads total.
// ---------------------------------------------------------------------------
__global__ __launch_bounds__(256)
void pack_w_kernel(const float* __restrict__ W, _Float16* __restrict__ wp) {
    const int tid  = blockIdx.x * 256 + threadIdx.x;  // [0, 8*16*32)
    const int kc   = tid >> 9;          // K-chunk 0..7
    const int rest = tid & 511;
    const int nt   = rest >> 5;         // N-tile 0..15
    const int lane = rest & 31;
    const int n    = nt * 16 + (lane & 15);
    const int kb   = kc * 32 + (lane >> 4) * 8;
    _Float16* dst = wp + ((size_t)(kc * 16 + nt) * 32 + lane) * 16;
    const float* src = W + (size_t)n * Vv + kb;
#pragma unroll
    for (int e = 0; e < 16; ++e)
        dst[e] = (_Float16)src[e + (e >= 8 ? 8 : 0)];
}

// ---------------------------------------------------------------------------
// WMMA GEMM:  out[m][o] = sum_v pm[m][v] * W[o][v] + bias[o]
// grid.x = MROWS/16 = 4095 blocks, 128 threads (4 waves).
// Each wave: 16 rows x 64 cols (4 x v8f accumulators), K unrolled 8x32.
// ---------------------------------------------------------------------------
__global__ __launch_bounds__(128)
void gemm_wmma_kernel(const _Float16* __restrict__ pm,
                      const _Float16* __restrict__ wp,
                      const float* __restrict__ bias,
                      float* __restrict__ out) {
    const int lane = threadIdx.x & 31;
    const int wave = threadIdx.x >> 5;     // 0..3 -> column group
    const int half = lane >> 4;            // 0/1
    const int mlow = lane & 15;
    const int r0 = blockIdx.x * 16;

    // A-fragment source row for this lane (ISA 16-bit A 16x32 layout:
    // elems 0..7 -> K = half*8 + e, elems 8..15 -> K = 16 + half*8 + e).
    const _Float16* aRow = pm + (size_t)(r0 + mlow) * Vv;

    v8f acc[4] = {};

#pragma unroll
    for (int kc = 0; kc < 8; ++kc) {
        const int k0 = kc * 32;
        const v8h lo = *(const v8h*)(aRow + k0 + half * 8);
        const v8h hi = *(const v8h*)(aRow + k0 + half * 8 + 16);
        v16h a;
#pragma unroll
        for (int e = 0; e < 8; ++e) { a[e] = lo[e]; a[e + 8] = hi[e]; }

        if (kc < 7)
            __builtin_prefetch(aRow + k0 + 32, 0, 1);   // next K-chunk of A

#pragma unroll
        for (int j = 0; j < 4; ++j) {
            const int nt = wave * 4 + j;                // global N-tile index
            const v16h bf = *(const v16h*)(wp + ((size_t)(kc * 16 + nt) * 32 + lane) * 16);
            acc[j] = __builtin_amdgcn_wmma_f32_16x16x32_f16(
                false, a, false, bf, (short)0, acc[j], false, false);
        }
    }

    // C/D layout: lane half selects M offset 8; VGPR i -> M = half*8 + i,
    // N = lane%16.  Fuse bias add into the store.
#pragma unroll
    for (int j = 0; j < 4; ++j) {
        const int col = wave * 64 + j * 16 + mlow;
        const float bs = bias[col];
        const int mb = r0 + half * 8;
        float* orow = out + (size_t)mb * Vv + col;
#pragma unroll
        for (int i = 0; i < 8; ++i)
            orow[(size_t)i * Vv] = acc[j][i] + bs;
    }
}

// ---------------------------------------------------------------------------
// Launch.  Inputs (setup_inputs order): 0=times(unused), 1=codes, 2=lengths
// (unused), 3=W, 4=b.  Workspace: pm (32MB f16) | wp (128KB f16) | cmx (1MB).
// ---------------------------------------------------------------------------
extern "C" void kernel_launch(void* const* d_in, const int* in_sizes, int n_in,
                              void* d_out, int out_size, void* d_ws, size_t ws_size,
                              hipStream_t stream) {
    const float* codes = (const float*)d_in[1];
    const float* W     = (const float*)d_in[3];
    const float* bias  = (const float*)d_in[4];
    float* out = (float*)d_out;

    char* ws = (char*)d_ws;
    _Float16* pm = (_Float16*)ws;                                  // 33,546,240 B
    _Float16* wp = (_Float16*)(ws + (size_t)MROWS * Vv * 2);       //    131,072 B
    float*   cmx = (float*)(ws + (size_t)MROWS * Vv * 2 + 131072); //  1,048,576 B

    chunkmax_kernel  <<<Bb * NCH, 256, 0, stream>>>(codes, cmx);
    carry_scan_kernel<<<16,       256, 0, stream>>>(cmx);
    prefix_kernel    <<<Bb * NCH, 256, 0, stream>>>(codes, cmx, pm);
    pack_w_kernel    <<<16,       256, 0, stream>>>(W, wp);
    gemm_wmma_kernel <<<MROWS / 16, 128, 0, stream>>>(pm, wp, bias, out);
}
